// PanguProMoEDecoderLayer_76802605187406
// MI455X (gfx1250) — matrix operand order repaired and checked by reading, not compiled, not run on hardware
//
#include <hip/hip_runtime.h>
#include <stdint.h>

// ---- model constants ----
#define T_SEQ 2048
#define HIDN  2048
#define NH    16
#define KVH   4
#define DH    128
#define VCHN  128
#define NE    8
#define EHID  512
#define SHID  1024
#define QKV_N (NH*DH + KVH*DH + KVH*VCHN)   // 3072

typedef __attribute__((ext_vector_type(16))) __bf16 bf16x16;
typedef __attribute__((ext_vector_type(8)))  float  f32x8;

union FragU { uint32_t u[8]; bf16x16 v; };

__device__ inline f32x8 zero8() {
  f32x8 z = {0.f,0.f,0.f,0.f,0.f,0.f,0.f,0.f};
  return z;
}
__device__ inline uint16_t f2bf(float f) {   // round-to-nearest-even
  uint32_t u = __float_as_uint(f);
  u += 0x7FFFu + ((u >> 16) & 1u);
  return (uint16_t)(u >> 16);
}
__device__ inline float bf2f(uint16_t h) { return __uint_as_float(((uint32_t)h) << 16); }

// ---- gfx1250 async global->LDS path (guarded: falls back to plain loads) ----
#if defined(__HIP_DEVICE_COMPILE__) && defined(__gfx1250__)
#if __has_builtin(__builtin_amdgcn_global_load_async_to_lds_b128)
#define ASYNC_LDS 1
// Parameter type per hipcc diagnostic: 'int __attribute__((vector_size(16))) __device__ *'
typedef int i32x4g __attribute__((vector_size(16)));
typedef __attribute__((address_space(1))) i32x4g* as1_v4i;   // global src
typedef __attribute__((address_space(3))) i32x4g* as3_v4i;   // LDS dst
#endif
#endif

__device__ inline void wait_async_then_nothing() {
#if defined(ASYNC_LDS)
#if __has_builtin(__builtin_amdgcn_s_wait_asynccnt)
  __builtin_amdgcn_s_wait_asynccnt(0);
#else
  asm volatile("s_wait_asynccnt 0x0" ::: "memory");
#endif
#endif
}

// ---------------- elementwise kernels ----------------
__global__ void f32_to_bf16_kernel(const float* __restrict__ s, uint16_t* __restrict__ d, int n) {
  for (int i = blockIdx.x * blockDim.x + threadIdx.x; i < n; i += gridDim.x * blockDim.x)
    d[i] = f2bf(s[i]);
}

__global__ void silu_mul_kernel(const float* __restrict__ a, const float* __restrict__ b,
                                uint16_t* __restrict__ o, int n) {
  for (int i = blockIdx.x * blockDim.x + threadIdx.x; i < n; i += gridDim.x * blockDim.x) {
    float g = a[i];
    g = g / (1.0f + __expf(-g));       // silu
    o[i] = f2bf(g * b[i]);
  }
}

__global__ void add_kernel(float* __restrict__ o, const float* __restrict__ a, int n) {
  for (int i = blockIdx.x * blockDim.x + threadIdx.x; i < n; i += gridDim.x * blockDim.x)
    o[i] += a[i];
}

// rmsnorm(x,w) -> bf16  (one block per row)
__global__ void rmsnorm_bf16_kernel(const float* __restrict__ x, const float* __restrict__ w,
                                    uint16_t* __restrict__ out, int ncol) {
  int row = blockIdx.x;
  const float* xr = x + (size_t)row * ncol;
  __shared__ float red[256];
  float ss = 0.f;
  for (int c = threadIdx.x; c < ncol; c += 256) { float v = xr[c]; ss += v * v; }
  red[threadIdx.x] = ss;
  __syncthreads();
  for (int s = 128; s > 0; s >>= 1) {
    if ((int)threadIdx.x < s) red[threadIdx.x] += red[threadIdx.x + s];
    __syncthreads();
  }
  float inv = rsqrtf(red[0] / (float)ncol + 1e-6f);
  for (int c = threadIdx.x; c < ncol; c += 256)
    out[(size_t)row * ncol + c] = f2bf(xr[c] * inv * w[c]);
}

// ---------------- WMMA GEMM (double-buffered LDS, templated epilogue) ----------------
// C = A(bf16 MxK) @ B(bf16 KxN)  [+bias[n]] [+resid[m][n]] [*rowscale[m*rs_stride]] [+=]
// Requires: M % 128 == 0, N % 128 == 0, K % 32 == 0 (true for all call sites).
#define TM 128
#define TN 128
#define TK 32

template <bool BIAS, bool RESID, bool SCALE, bool ACCUM>
__global__ __launch_bounds__(256) void gemm_bf16_kernel(
    const uint16_t* __restrict__ A, const uint16_t* __restrict__ B, float* __restrict__ C,
    int M, int N, int K,
    const float* __restrict__ bias, const float* __restrict__ resid,
    const float* __restrict__ rowscale, int rs_stride) {
  __shared__ uint16_t As[2][TM][TK + 8];
  __shared__ uint16_t Bt[2][TN][TK + 8];     // B stored transposed: Bt[n][k]
  const int tid   = threadIdx.x;
  const int lane  = tid & 31;
  const int wave  = tid >> 5;
  const int l16   = lane & 15;
  const int lhalf = lane >> 4;
  const int wm = wave >> 1, wn = wave & 1;
  const int bm = blockIdx.y * TM, bn = blockIdx.x * TN;

  f32x8 acc[2][4];
#pragma unroll
  for (int i = 0; i < 2; i++)
#pragma unroll
    for (int j = 0; j < 4; j++) acc[i][j] = zero8();

  const int ar  = tid >> 1, ah = (tid & 1) * 16;  // A staging: row, 16-col half
  const int bk  = tid >> 3, bn0 = (tid & 7) * 16; // B staging: k-row, 16-col group

  auto stageA = [&](int buf, int kk) {
    const uint16_t* src = A + (size_t)(bm + ar) * K + kk + ah;
    uint16_t* dst = &As[buf][ar][ah];
#if defined(ASYNC_LDS)
    __builtin_amdgcn_global_load_async_to_lds_b128((as1_v4i)(void*)src, (as3_v4i)(void*)dst, 0, 0);
    __builtin_amdgcn_global_load_async_to_lds_b128((as1_v4i)(void*)(src + 8), (as3_v4i)(void*)(dst + 8), 0, 0);
#else
    const uint32_t* s32 = (const uint32_t*)src;
    uint32_t* d32 = (uint32_t*)dst;
#pragma unroll
    for (int i = 0; i < 8; i++) d32[i] = s32[i];
#endif
  };
  auto stageB = [&](int buf, int kk) {
    const uint16_t* src = B + (size_t)(kk + bk) * N + bn + bn0;
#pragma unroll
    for (int i = 0; i < 16; i++) Bt[buf][bn0 + i][bk] = src[i];
  };

  const int nk = K / TK;
  stageA(0, 0);
  stageB(0, 0);
  wait_async_then_nothing();
  __syncthreads();

  for (int step = 0; step < nk; step++) {
    const int cur = step & 1, nxt = cur ^ 1;
    if (step + 1 < nk) {               // stage next tile while computing current
      stageA(nxt, (step + 1) * TK);
      stageB(nxt, (step + 1) * TK);
    }
    if (step + 2 < nk) {               // L2 prefetch two steps ahead (global_prefetch_b8)
      __builtin_prefetch(A + (size_t)(bm + ar) * K + (step + 2) * TK + ah, 0, 1);
      __builtin_prefetch(B + (size_t)((step + 2) * TK + bk) * N + bn + bn0, 0, 1);
    }

    FragU a[2];
#pragma unroll
    for (int mi = 0; mi < 2; mi++) {
      // A 16x32 layout: lane row=l16; elems 0..15 = K lhalf*16+i
      const uint32_t* p0 = (const uint32_t*)&As[cur][wm * 32 + mi * 16 + l16][lhalf * 16];
      const uint32_t* p1 = (const uint32_t*)&As[cur][wm * 32 + mi * 16 + l16][lhalf * 16 + 8];
#pragma unroll
      for (int i = 0; i < 4; i++) { a[mi].u[i] = p0[i]; a[mi].u[4 + i] = p1[i]; }
    }
#pragma unroll
    for (int ni = 0; ni < 4; ni++) {
      // B 32x16 layout: lane col=l16; elems 0..15 = K lhalf*16 + i
      FragU b;
      const uint32_t* p = (const uint32_t*)&Bt[cur][wn * 64 + ni * 16 + l16][lhalf * 16];
#pragma unroll
      for (int i = 0; i < 8; i++) b.u[i] = p[i];
#pragma unroll
      for (int mi = 0; mi < 2; mi++)
        acc[mi][ni] = __builtin_amdgcn_wmma_f32_16x16x32_bf16(
            false, a[mi].v, false, b.v, (short)0, acc[mi][ni], false, false);
    }
    wait_async_then_nothing();   // wave's async stores to LDS done before barrier
    __syncthreads();
  }

#pragma unroll
  for (int mi = 0; mi < 2; mi++)
#pragma unroll
    for (int ni = 0; ni < 4; ni++)
#pragma unroll
      for (int r = 0; r < 8; r++) {
        int row = bm + wm * 32 + mi * 16 + lhalf * 8 + r;  // C/D: VGPR r -> M=r (+8 hi half)
        int col = bn + wn * 64 + ni * 16 + l16;
        float v = acc[mi][ni][r];
        if (BIAS)  v += bias[col];
        if (RESID) v += resid[(size_t)row * N + col];
        if (SCALE) v *= rowscale[(size_t)row * rs_stride];
        size_t o = (size_t)row * N + col;
        if (ACCUM) C[o] += v; else C[o] = v;
      }
}

// ---------------- k-rmsnorm + RoPE + split qkv -> bf16 q/k/v ----------------
__global__ __launch_bounds__(128) void rope_split_kernel(
    const float* __restrict__ qkv, const int* __restrict__ pos, const float* __restrict__ k_ln_w,
    uint16_t* __restrict__ qo, uint16_t* __restrict__ ko, uint16_t* __restrict__ vo) {
  const int t = blockIdx.x, hh = blockIdx.y, d = threadIdx.x;
  const float* row = qkv + (size_t)t * QKV_N;
  __shared__ float red[128];
  const float p = (float)pos[t];

  if (hh < NH) {                                  // query head: rope on last 64 dims
    const float* hp = row + hh * DH;
    float outv;
    if (d < 64) outv = hp[d];
    else {
      int i = (d - 64) & 31;
      float invf = __powf(10000.0f, -((float)i) / 32.0f);
      float ang = p * invf, c = __cosf(ang), s = __sinf(ang);
      float x1 = hp[64 + i], x2 = hp[96 + i];
      outv = (d < 96) ? (x1 * c - x2 * s) : (x1 * s + x2 * c);
    }
    qo[(size_t)t * (NH * DH) + hh * DH + d] = f2bf(outv);
  } else if (hh < NH + KVH) {                      // key head: rmsnorm then rope
    int kh = hh - NH;
    const float* hp = row + NH * DH + kh * DH;
    float v = hp[d];
    red[d] = v * v;
    __syncthreads();
    for (int s = 64; s > 0; s >>= 1) { if (d < s) red[d] += red[d + s]; __syncthreads(); }
    float inv = rsqrtf(red[0] / (float)DH + 1e-6f);
    float outv;
    if (d < 64) outv = v * inv * k_ln_w[d];
    else {
      int i = (d - 64) & 31;
      float invf = __powf(10000.0f, -((float)i) / 32.0f);
      float ang = p * invf, c = __cosf(ang), s = __sinf(ang);
      float x1 = hp[64 + i] * inv * k_ln_w[64 + i];
      float x2 = hp[96 + i] * inv * k_ln_w[96 + i];
      outv = (d < 96) ? (x1 * c - x2 * s) : (x1 * s + x2 * c);
    }
    ko[(size_t)t * (KVH * DH) + kh * DH + d] = f2bf(outv);
  } else {                                         // value head: passthrough
    int vh = hh - NH - KVH;
    vo[(size_t)t * (KVH * VCHN) + vh * VCHN + d] =
        f2bf(row[NH * DH + KVH * DH + vh * VCHN + d]);
  }
}

// ---------------- flash attention (causal, GQA 4:1), WMMA QK^T and PV ----------------
__global__ __launch_bounds__(128) void flash_attn_kernel(
    const uint16_t* __restrict__ q, const uint16_t* __restrict__ k, const uint16_t* __restrict__ v,
    float* __restrict__ out) {
  __shared__ uint16_t Ks[32][128];      // K rows (== Bt layout for QK^T)
  __shared__ uint16_t Vt[128][36];      // V transposed: Vt[channel][key]
  __shared__ uint16_t Ps[4][16][32];    // per-wave P tile for A-frag conversion

  const int tid = threadIdx.x;
  const int lane = tid & 31, w = tid >> 5;
  const int l16 = lane & 15, lhalf = lane >> 4;
  const int h = blockIdx.y, hkv = h >> 2;
  const int qbase = blockIdx.x * 64;
  const int rbase = qbase + w * 16;

  // Q A-fragments for 4 chunks of d (32 each)
  FragU aq[4];
  {
    const uint16_t* qp = q + (size_t)(rbase + l16) * (NH * DH) + h * DH;
#pragma unroll
    for (int c = 0; c < 4; c++) {
      const uint32_t* p0 = (const uint32_t*)(qp + c * 32 + lhalf * 16);
      const uint32_t* p1 = (const uint32_t*)(qp + c * 32 + lhalf * 16 + 8);
#pragma unroll
      for (int i = 0; i < 4; i++) { aq[c].u[i] = p0[i]; aq[c].u[4 + i] = p1[i]; }
    }
  }

  float mrow[8], lrow[8];
  f32x8 acc_o[8];
#pragma unroll
  for (int r = 0; r < 8; r++) { mrow[r] = -3.0e38f; lrow[r] = 0.f; }
#pragma unroll
  for (int nc = 0; nc < 8; nc++) acc_o[nc] = zero8();

  const float scale = 0.08838834764831845f;  // 128^-0.5
  const int jmax = (qbase + 63) >> 5;
  for (int jb = 0; jb <= jmax; jb++) {
    const int kb0 = jb * 32;
    {   // cooperative K/V staging: 32 keys x 128
      int key = tid >> 2, seg = (tid & 3) * 32;
      const uint32_t* src = (const uint32_t*)(k + (size_t)(kb0 + key) * (KVH * DH) + hkv * DH + seg);
      uint32_t* dst = (uint32_t*)&Ks[key][seg];
#pragma unroll
      for (int i = 0; i < 16; i++) dst[i] = src[i];
      const uint16_t* vsrc = v + (size_t)(kb0 + key) * (KVH * VCHN) + hkv * VCHN + seg;
#pragma unroll
      for (int i = 0; i < 32; i++) Vt[seg + i][key] = vsrc[i];
      if (jb < jmax) {   // prefetch next K/V block into L2 (global_prefetch_b8)
        __builtin_prefetch(k + (size_t)(kb0 + 32 + key) * (KVH * DH) + hkv * DH + seg, 0, 1);
        __builtin_prefetch(v + (size_t)(kb0 + 32 + key) * (KVH * VCHN) + hkv * VCHN + seg, 0, 1);
      }
    }
    __syncthreads();

    if (kb0 <= rbase + 15) {
      // S = Q @ K^T  (two 16-key subtiles)
      f32x8 s0 = zero8(), s1 = zero8();
#pragma unroll
      for (int c = 0; c < 4; c++) {
        FragU bk;
        const uint32_t* p = (const uint32_t*)&Ks[l16][c * 32 + lhalf * 16];
#pragma unroll
        for (int i = 0; i < 8; i++) bk.u[i] = p[i];
        s0 = __builtin_amdgcn_wmma_f32_16x16x32_bf16(false, aq[c].v, false, bk.v, (short)0, s0, false, false);
        p = (const uint32_t*)&Ks[16 + l16][c * 32 + lhalf * 16];
#pragma unroll
        for (int i = 0; i < 8; i++) bk.u[i] = p[i];
        s1 = __builtin_amdgcn_wmma_f32_16x16x32_bf16(false, aq[c].v, false, bk.v, (short)0, s1, false, false);
      }
      // online softmax per row (rows striped over accumulator elements)
#pragma unroll
      for (int r = 0; r < 8; r++) {
        int row = rbase + lhalf * 8 + r;
        int c0 = kb0 + l16, c1 = kb0 + 16 + l16;
        float f0 = s0[r] * scale; if (c0 > row) f0 = -1e30f;
        float f1 = s1[r] * scale; if (c1 > row) f1 = -1e30f;
        float tmax = fmaxf(f0, f1);
#pragma unroll
        for (int m = 1; m < 16; m <<= 1) tmax = fmaxf(tmax, __shfl_xor(tmax, m, 32));
        float mnew = fmaxf(mrow[r], tmax);
        float corr = __expf(mrow[r] - mnew);
        float p0 = __expf(f0 - mnew), p1 = __expf(f1 - mnew);
        float ts = p0 + p1;
#pragma unroll
        for (int m = 1; m < 16; m <<= 1) ts += __shfl_xor(ts, m, 32);
        lrow[r] = lrow[r] * corr + ts;
        mrow[r] = mnew;
#pragma unroll
        for (int nc = 0; nc < 8; nc++) acc_o[nc][r] *= corr;
        int rl = lhalf * 8 + r;
        Ps[w][rl][l16]      = f2bf(p0);
        Ps[w][rl][16 + l16] = f2bf(p1);
      }
      // wave-local LDS RAW: make P stores visible before A-frag reads
      asm volatile("s_wait_dscnt 0x0" ::: "memory");
      FragU ap;
      {
        const uint32_t* p0 = (const uint32_t*)&Ps[w][l16][lhalf * 16];
        const uint32_t* p1 = (const uint32_t*)&Ps[w][l16][lhalf * 16 + 8];
#pragma unroll
        for (int i = 0; i < 4; i++) { ap.u[i] = p0[i]; ap.u[4 + i] = p1[i]; }
      }
      // O += P @ V
#pragma unroll
      for (int nc = 0; nc < 8; nc++) {
        FragU bv;
        const uint32_t* p = (const uint32_t*)&Vt[nc * 16 + l16][lhalf * 16];
#pragma unroll
        for (int i = 0; i < 8; i++) bv.u[i] = p[i];
        acc_o[nc] = __builtin_amdgcn_wmma_f32_16x16x32_bf16(
            false, ap.v, false, bv.v, (short)0, acc_o[nc], false, false);
      }
    }
    __syncthreads();
  }

#pragma unroll
  for (int r = 0; r < 8; r++) {
    int row = rbase + lhalf * 8 + r;
    float inv = 1.0f / lrow[r];
#pragma unroll
    for (int nc = 0; nc < 8; nc++)
      out[(size_t)row * (NH * VCHN) + h * VCHN + nc * 16 + l16] = acc_o[nc][r] * inv;
  }
}

// ---------------- router: softmax(h2 @ W) -> top2 -> normalized combine[T][E] ----------------
__global__ __launch_bounds__(64) void router_kernel(const uint16_t* __restrict__ h2,
                                                    const float* __restrict__ rw,
                                                    float* __restrict__ combine) {
  int t = blockIdx.x, tid = threadIdx.x;
  __shared__ float red[64][NE];
  float acc[NE];
#pragma unroll
  for (int e = 0; e < NE; e++) acc[e] = 0.f;
  for (int d = tid; d < HIDN; d += 64) {
    float hv = bf2f(h2[(size_t)t * HIDN + d]);
#pragma unroll
    for (int e = 0; e < NE; e++) acc[e] += hv * rw[d * NE + e];
  }
#pragma unroll
  for (int e = 0; e < NE; e++) red[tid][e] = acc[e];
  __syncthreads();
  for (int s = 32; s > 0; s >>= 1) {
    if (tid < s)
#pragma unroll
      for (int e = 0; e < NE; e++) red[tid][e] += red[tid + s][e];
    __syncthreads();
  }
  if (tid == 0) {
    float lg[NE], mx = -3.0e38f;
#pragma unroll
    for (int e = 0; e < NE; e++) { lg[e] = red[0][e]; mx = fmaxf(mx, lg[e]); }
    float sum = 0.f;
#pragma unroll
    for (int e = 0; e < NE; e++) { lg[e] = __expf(lg[e] - mx); sum += lg[e]; }
#pragma unroll
    for (int e = 0; e < NE; e++) lg[e] /= sum;
    int i1 = 0;
#pragma unroll
    for (int e = 1; e < NE; e++) if (lg[e] > lg[i1]) i1 = e;
    int i2 = (i1 == 0) ? 1 : 0;
#pragma unroll
    for (int e = 0; e < NE; e++) if (e != i1 && lg[e] > lg[i2]) i2 = e;
    float ws = lg[i1] + lg[i2];
#pragma unroll
    for (int e = 0; e < NE; e++)
      combine[(size_t)t * NE + e] = (e == i1) ? lg[i1] / ws : (e == i2) ? lg[i2] / ws : 0.f;
  }
}

// ---------------- host-side orchestration ----------------
static inline void launch_gemm(dim3 grid, hipStream_t s,
                               const uint16_t* A, const uint16_t* B, float* C,
                               int M, int N, int K,
                               const float* bias, const float* resid,
                               const float* rowscale, int rs_stride) {
  if (bias)
    gemm_bf16_kernel<true, false, false, false><<<grid, 256, 0, s>>>(A, B, C, M, N, K, bias, nullptr, nullptr, 0);
  else if (resid)
    gemm_bf16_kernel<false, true, false, false><<<grid, 256, 0, s>>>(A, B, C, M, N, K, nullptr, resid, nullptr, 0);
  else if (rowscale)   // rowscale implies accumulate (expert down-proj)
    gemm_bf16_kernel<false, false, true, true><<<grid, 256, 0, s>>>(A, B, C, M, N, K, nullptr, nullptr, rowscale, rs_stride);
  else
    gemm_bf16_kernel<false, false, false, false><<<grid, 256, 0, s>>>(A, B, C, M, N, K, nullptr, nullptr, nullptr, 0);
}

extern "C" void kernel_launch(void* const* d_in, const int* in_sizes, int n_in,
                              void* d_out, int out_size, void* d_ws, size_t ws_size,
                              hipStream_t stream) {
  (void)in_sizes; (void)n_in; (void)out_size; (void)ws_size;
  const float* x        = (const float*)d_in[0];
  const int*   pos      = (const int*)d_in[1];
  const float* ln1_w    = (const float*)d_in[2];
  const float* qkv_w    = (const float*)d_in[3];
  const float* qkv_b    = (const float*)d_in[4];
  const float* k_ln_w   = (const float*)d_in[5];
  const float* o_w      = (const float*)d_in[6];
  const float* ln2_w    = (const float*)d_in[7];
  const float* router_w = (const float*)d_in[8];
  const float* w1       = (const float*)d_in[9];
  const float* w2       = (const float*)d_in[10];
  const float* w3       = (const float*)d_in[11];
  const float* sw1      = (const float*)d_in[12];
  const float* sw2      = (const float*)d_in[13];
  const float* sw3      = (const float*)d_in[14];
  float* out = (float*)d_out;

  char* ws = (char*)d_ws;
  size_t off = 0;
  auto alloc = [&](size_t bytes) -> void* {
    void* p = ws + off;
    off = (off + bytes + 255) & ~(size_t)255;
    return p;
  };

  // bf16 weight copies (rebuilt every call: deterministic, no cached state)
  uint16_t* bqkv = (uint16_t*)alloc((size_t)HIDN * QKV_N * 2);
  uint16_t* bo   = (uint16_t*)alloc((size_t)HIDN * HIDN * 2);
  uint16_t* bsw1 = (uint16_t*)alloc((size_t)HIDN * SHID * 2);
  uint16_t* bsw2 = (uint16_t*)alloc((size_t)SHID * HIDN * 2);
  uint16_t* bsw3 = (uint16_t*)alloc((size_t)HIDN * SHID * 2);
  uint16_t* bw1  = (uint16_t*)alloc((size_t)NE * HIDN * EHID * 2);
  uint16_t* bw2  = (uint16_t*)alloc((size_t)NE * EHID * HIDN * 2);
  uint16_t* bw3  = (uint16_t*)alloc((size_t)NE * HIDN * EHID * 2);
  // activations
  uint16_t* hbf   = (uint16_t*)alloc((size_t)T_SEQ * HIDN * 2);
  float*    qkvf  = (float*)   alloc((size_t)T_SEQ * QKV_N * 4);
  uint16_t* qb    = (uint16_t*)alloc((size_t)T_SEQ * NH * DH * 2);
  uint16_t* kb    = (uint16_t*)alloc((size_t)T_SEQ * KVH * DH * 2);
  uint16_t* vb    = (uint16_t*)alloc((size_t)T_SEQ * KVH * VCHN * 2);
  float*    attnf = (float*)   alloc((size_t)T_SEQ * HIDN * 4);
  uint16_t* attnb = (uint16_t*)alloc((size_t)T_SEQ * HIDN * 2);
  uint16_t* h2b   = (uint16_t*)alloc((size_t)T_SEQ * HIDN * 2);
  float*    comb  = (float*)   alloc((size_t)T_SEQ * NE * 4);
  float*    g1    = (float*)   alloc((size_t)T_SEQ * SHID * 4);
  float*    g3    = (float*)   alloc((size_t)T_SEQ * SHID * 4);
  uint16_t* gb    = (uint16_t*)alloc((size_t)T_SEQ * SHID * 2);
  float*    moe   = (float*)   alloc((size_t)T_SEQ * HIDN * 4);
  float*    e1    = (float*)   alloc((size_t)T_SEQ * EHID * 4);
  float*    e3    = (float*)   alloc((size_t)T_SEQ * EHID * 4);
  uint16_t* eb    = (uint16_t*)alloc((size_t)T_SEQ * EHID * 2);

  auto cvt = [&](const float* s, uint16_t* d, size_t n) {
    int blocks = (int)((n + 255) / 256);
    if (blocks > 4096) blocks = 4096;
    f32_to_bf16_kernel<<<blocks, 256, 0, stream>>>(s, d, (int)n);
  };
  cvt(qkv_w, bqkv, (size_t)HIDN * QKV_N);
  cvt(o_w,   bo,   (size_t)HIDN * HIDN);
  cvt(sw1,   bsw1, (size_t)HIDN * SHID);
  cvt(sw2,   bsw2, (size_t)SHID * HIDN);
  cvt(sw3,   bsw3, (size_t)HIDN * SHID);
  cvt(w1,    bw1,  (size_t)NE * HIDN * EHID);
  cvt(w2,    bw2,  (size_t)NE * EHID * HIDN);
  cvt(w3,    bw3,  (size_t)NE * HIDN * EHID);

  // h = rmsnorm(x, ln1_w)
  rmsnorm_bf16_kernel<<<T_SEQ, 256, 0, stream>>>(x, ln1_w, hbf, HIDN);
  // qkv = h @ qkv_w + qkv_b
  launch_gemm(dim3(QKV_N / TN, T_SEQ / TM), stream, hbf, bqkv, qkvf,
              T_SEQ, QKV_N, HIDN, qkv_b, nullptr, nullptr, 0);
  // k-rmsnorm + rope + split to bf16
  rope_split_kernel<<<dim3(T_SEQ, NH + 2 * KVH), 128, 0, stream>>>(qkvf, pos, k_ln_w, qb, kb, vb);
  // causal flash attention
  flash_attn_kernel<<<dim3(T_SEQ / 64, NH), 128, 0, stream>>>(qb, kb, vb, attnf);
  cvt(attnf, attnb, (size_t)T_SEQ * HIDN);
  // x1 = x + attn @ o_w   (residual fused in epilogue, written to d_out)
  launch_gemm(dim3(HIDN / TN, T_SEQ / TM), stream, attnb, bo, out,
              T_SEQ, HIDN, HIDN, nullptr, x, nullptr, 0);
  // h2 = rmsnorm(x1, ln2_w)
  rmsnorm_bf16_kernel<<<T_SEQ, 256, 0, stream>>>(out, ln2_w, h2b, HIDN);
  // router -> combine weights
  router_kernel<<<T_SEQ, 64, 0, stream>>>(h2b, router_w, comb);
  // shared expert: moe = swiglu(h2, sw1, sw3, sw2)
  launch_gemm(dim3(SHID / TN, T_SEQ / TM), stream, h2b, bsw1, g1,
              T_SEQ, SHID, HIDN, nullptr, nullptr, nullptr, 0);
  launch_gemm(dim3(SHID / TN, T_SEQ / TM), stream, h2b, bsw3, g3,
              T_SEQ, SHID, HIDN, nullptr, nullptr, nullptr, 0);
  silu_mul_kernel<<<4096, 256, 0, stream>>>(g1, g3, gb, T_SEQ * SHID);
  launch_gemm(dim3(HIDN / TN, T_SEQ / TM), stream, gb, bsw2, moe,
              T_SEQ, HIDN, SHID, nullptr, nullptr, nullptr, 0);
  // routed experts (dense, per-row combine weight fused into down-proj epilogue)
  for (int e = 0; e < NE; e++) {
    launch_gemm(dim3(EHID / TN, T_SEQ / TM), stream, h2b, bw1 + (size_t)e * HIDN * EHID, e1,
                T_SEQ, EHID, HIDN, nullptr, nullptr, nullptr, 0);
    launch_gemm(dim3(EHID / TN, T_SEQ / TM), stream, h2b, bw3 + (size_t)e * HIDN * EHID, e3,
                T_SEQ, EHID, HIDN, nullptr, nullptr, nullptr, 0);
    silu_mul_kernel<<<4096, 256, 0, stream>>>(e1, e3, eb, T_SEQ * EHID);
    launch_gemm(dim3(HIDN / TN, T_SEQ / TM), stream, eb, bw2 + (size_t)e * EHID * HIDN, moe,
                T_SEQ, HIDN, EHID, nullptr, nullptr, comb + e, NE);
  }
  // final: out = x1 + moe
  add_kernel<<<4096, 256, 0, stream>>>(out, moe, T_SEQ * HIDN);
}
